// AtomEmbedding_9174050144965
// MI455X (gfx1250) — matrix operand order repaired and checked by reading, not compiled, not run on hardware
//
#include <hip/hip_runtime.h>
#include <hip/hip_bf16.h>
#include <stdint.h>

typedef _Float16 half_t;
typedef __attribute__((ext_vector_type(16))) _Float16 v16h;
typedef __attribute__((ext_vector_type(8)))  float    v8f;

static __device__ __forceinline__ v8f wmma16(v16h a, v16h b, v8f c) {
  // D = A(16x32 f16) * B(32x16 f16) + C(16x16 f32)
  return __builtin_amdgcn_wmma_f32_16x16x32_f16(false, a, false, b, (short)0, c,
                                                false, false);
}

// Async global->LDS DMA (CDNA5): per-lane 16B, tracked by ASYNCcnt.
// LDS dest offset comes from a VGPR; global address from a VGPR pair.
static __device__ __forceinline__ void async_copy16(uint32_t lds_off,
                                                    const void* gptr) {
  asm volatile("global_load_async_to_lds_b128 %0, %1, off"
               :
               : "v"(lds_off), "v"((uint64_t)(uintptr_t)gptr)
               : "memory");
}
static __device__ __forceinline__ void wait_async0() {
  asm volatile("s_wait_asynccnt 0" ::: "memory");
}
// cooperative async stage of `halves` f16 (multiple of 512) into LDS
static __device__ __forceinline__ void stage_async(half_t* dst,
                                                   const half_t* src,
                                                   int halves, int tid,
                                                   int nthreads) {
  for (int idx = tid * 8; idx < halves; idx += nthreads * 8)
    async_copy16((uint32_t)(uintptr_t)&dst[idx], src + idx);
  wait_async0();
}

static __device__ __forceinline__ int enc_f(float f) {
  int i = __float_as_int(f);
  return i >= 0 ? i : (i ^ 0x7FFFFFFF);
}
static __device__ __forceinline__ float dec_f(int k) {
  int i = k >= 0 ? k : (k ^ 0x7FFFFFFF);
  return __int_as_float(i);
}
static __device__ __forceinline__ float lrelu(float v) {
  return v > 0.f ? v : 0.01f * v;
}
static __device__ __forceinline__ float sigm(float v) {
  return 1.f / (1.f + __expf(-v));
}

// ---------------------------------------------------------------------------
// Pack fp32 weight W[N_out][K_in] into f16 WMMA B-fragment order:
// offset ((nt*ktCount + kt)*32 + lane)*16 + i  <->  B[k = kt*32 + (lane>>4)*16 + i]
//                                                    [n = nt*16 + (lane&15)]
// so each lane's 16 values (32B) are contiguous -> single-shot LDS b128 loads.
// ---------------------------------------------------------------------------
__global__ void pack_weight_kernel(const float* __restrict__ w,
                                   half_t* __restrict__ o,
                                   int K_in, int K_pad, int N_out) {
  int ktCount = K_pad >> 5;
  int total = (N_out >> 4) * ktCount * 512;
  int t = blockIdx.x * blockDim.x + threadIdx.x;
  if (t >= total) return;
  int i    = t & 15;
  int lane = (t >> 4) & 31;
  int r2   = t >> 9;
  int kt   = r2 % ktCount;
  int nt   = r2 / ktCount;
  int k = kt * 32 + ((lane >> 4) & 1) * 16 + i;
  int n = nt * 16 + (lane & 15);
  float v = (k < K_in) ? w[n * K_in + k] : 0.f;
  o[t] = (half_t)v;
}

// A-tile fragment-order store helper: element (row r in 0..15, k) of 16xK tile
static __device__ __forceinline__ int a_frag_off(int r, int k) {
  int kt = k >> 5, kk = k & 31;
  int lt = ((kk >> 4) & 1) * 16 + r;           // lane
  int j  = kk & 15;                            // element
  return (kt * 32 + lt) * 16 + j;
}

// ---------------------------------------------------------------------------
// Per-node GEMM: out[n_rows][N_out] = act(in[n_rows][K_in] @ Wp + bias)
// Block = 128 threads (4 waves), 16 rows per wave.
// ---------------------------------------------------------------------------
__global__ __launch_bounds__(128) void node_gemm_kernel(
    const float* __restrict__ in, int K_in, int K_pad,
    const half_t* __restrict__ Wp, int N_out,
    const float* __restrict__ bias,
    float* __restrict__ out, int n_rows, int act) {
  __shared__ alignas(32) half_t Wlds[8 * 4 * 512];   // <=8 ntiles x <=4 kt: 32KB
  __shared__ alignas(32) half_t Alds[4][4 * 512];    // per-wave A tile: 16KB
  const int tid = threadIdx.x, wave = tid >> 5, lane = tid & 31;
  const int ktCount = K_pad >> 5;
  const int row0 = blockIdx.x * 64 + wave * 16;

  for (int idx = lane; idx < 16 * K_pad; idx += 32) {
    int r = idx / K_pad, k = idx - r * K_pad;
    int grow = row0 + r;
    float v = 0.f;
    if (grow < n_rows && k < K_in) v = in[grow * K_in + k];
    Alds[wave][a_frag_off(r, k)] = (half_t)v;
  }

  const int half_id = lane >> 4, nl = lane & 15;
  const int ntTotal = N_out >> 4;

  for (int ntc = 0; ntc < ntTotal; ntc += 8) {
    int ntCnt = ntTotal - ntc; if (ntCnt > 8) ntCnt = 8;
    int chunkHalves = ntCnt * ktCount * 512;
    const half_t* src = Wp + (size_t)ntc * ktCount * 512;
    __syncthreads();
    stage_async(Wlds, src, chunkHalves, tid, 128);   // async DMA global->LDS
    __syncthreads();
    for (int ntl = 0; ntl < ntCnt; ++ntl) {
      float bv = bias[(ntc + ntl) * 16 + nl];
      v8f acc = {bv, bv, bv, bv, bv, bv, bv, bv};
      for (int kt = 0; kt < ktCount; ++kt) {
        v16h a = *(const v16h*)(&Alds[wave][(kt * 32 + lane) * 16]);
        v16h b = *(const v16h*)(&Wlds[((ntl * ktCount + kt) * 32 + lane) * 16]);
        acc = wmma16(a, b, acc);
      }
#pragma unroll
      for (int m = 0; m < 8; ++m) {
        int grow = row0 + m + half_id * 8;
        if (grow < n_rows) {
          float v = acc[m];
          if (act == 1) v = lrelu(v);
          out[(size_t)grow * N_out + (ntc + ntl) * 16 + nl] = v;
        }
      }
    }
  }
}

// ---------------------------------------------------------------------------
// Per-node dot with a 128-long weight vector: outp[n] = dot(v[n][:dim], aw)
// ---------------------------------------------------------------------------
__global__ __launch_bounds__(256) void node_dot_kernel(
    const float* __restrict__ v, const float* __restrict__ aw,
    int dim, float* __restrict__ outp, int n) {
  int wave = threadIdx.x >> 5, lane = threadIdx.x & 31;
  int node = blockIdx.x * 8 + wave;
  if (node >= n) return;
  float s = 0.f;
  for (int k = lane; k < dim; k += 32) s += v[(size_t)node * dim + k] * aw[k];
  s += __shfl_xor(s, 1, 32);
  s += __shfl_xor(s, 2, 32);
  s += __shfl_xor(s, 4, 32);
  s += __shfl_xor(s, 8, 32);
  s += __shfl_xor(s, 16, 32);
  if (lane == 0) outp[node] = s;
}

// ---------------------------------------------------------------------------
// Layer-0 edge attention logit:
//   u = [x[src](39) | edge_attr(10)] pad 64
//   xj = lrelu(u @ Wn + bn)  (16x128 tile via WMMA, never materialized)
//   evu[e] = lrelu( dot(xj, alignW[128:]) + p0[dst] + alignB )
// ---------------------------------------------------------------------------
__global__ __launch_bounds__(128) void edge_evu_l0_kernel(
    const float* __restrict__ X, const int* __restrict__ EI,
    const float* __restrict__ EA,
    const half_t* __restrict__ Wn,      // packed: 8 nt x 2 kt
    const float* __restrict__ bn,
    const float* __restrict__ alignW,   // [256]
    const float* __restrict__ alignB,   // [1]
    const float* __restrict__ p0,       // [N]
    float* __restrict__ evu, int E) {
  __shared__ alignas(32) half_t Wlds[8 * 2 * 512];   // 16KB
  __shared__ alignas(32) half_t Ulds[4][2 * 512];    // 8KB
  const int tid = threadIdx.x, wave = tid >> 5, lane = tid & 31;
  const int e0 = blockIdx.x * 64 + wave * 16;

  stage_async(Wlds, Wn, 8 * 2 * 512, tid, 128);      // async DMA global->LDS
  for (int idx = lane; idx < 16 * 64; idx += 32) {
    int r = idx >> 6, k = idx & 63;
    int e = e0 + r;
    float v = 0.f;
    if (e < E) {
      if (k < 39)      v = X[(size_t)EI[e] * 39 + k];
      else if (k < 49) v = EA[(size_t)e * 10 + (k - 39)];
    }
    Ulds[wave][a_frag_off(r, k)] = (half_t)v;
  }
  __syncthreads();

  const int half_id = lane >> 4, nl = lane & 15;
  float pf[8];
#pragma unroll
  for (int m = 0; m < 8; ++m) pf[m] = 0.f;

  for (int nt = 0; nt < 8; ++nt) {
    float bv = bn[nt * 16 + nl];
    v8f acc = {bv, bv, bv, bv, bv, bv, bv, bv};
#pragma unroll
    for (int kt = 0; kt < 2; ++kt) {
      v16h a = *(const v16h*)(&Ulds[wave][(kt * 32 + lane) * 16]);
      v16h b = *(const v16h*)(&Wlds[((nt * 2 + kt) * 32 + lane) * 16]);
      acc = wmma16(a, b, acc);
    }
    float aw2 = alignW[128 + nt * 16 + nl];
#pragma unroll
    for (int m = 0; m < 8; ++m) pf[m] += lrelu(acc[m]) * aw2;
  }
#pragma unroll
  for (int m = 0; m < 8; ++m) {
    pf[m] += __shfl_xor(pf[m], 1, 32);
    pf[m] += __shfl_xor(pf[m], 2, 32);
    pf[m] += __shfl_xor(pf[m], 4, 32);
    pf[m] += __shfl_xor(pf[m], 8, 32);
  }
  if (nl == 0) {
    float ab = alignB[0];
#pragma unroll
    for (int m = 0; m < 8; ++m) {
      int e = e0 + m + half_id * 8;
      if (e < E) evu[e] = lrelu(pf[m] + p0[EI[E + e]] + ab);
    }
  }
}

// Layer-1 logit: evu = lrelu(p[dst] + q[src] + ab)
__global__ void edge_evu_l1_kernel(const int* __restrict__ EI,
                                   const float* __restrict__ p,
                                   const float* __restrict__ q,
                                   const float* __restrict__ ab,
                                   float* __restrict__ evu, int E) {
  int e = blockIdx.x * blockDim.x + threadIdx.x;
  if (e < E) evu[e] = lrelu(p[EI[E + e]] + q[EI[e]] + ab[0]);
}

// ---------------------------------------------------------------------------
// Segment softmax pieces (keyed by src)
// ---------------------------------------------------------------------------
__global__ void init_seg_kernel(int* __restrict__ mkey, float* __restrict__ s, int n) {
  int t = blockIdx.x * blockDim.x + threadIdx.x;
  if (t < n) { mkey[t] = (int)0x80000000; s[t] = 0.f; }
}
__global__ void seg_max_kernel(const int* __restrict__ EI,
                               const float* __restrict__ evu,
                               int* __restrict__ mkey, int E) {
  int e = blockIdx.x * blockDim.x + threadIdx.x;
  if (e < E) atomicMax(&mkey[EI[e]], enc_f(evu[e]));
}
__global__ void seg_expsum_kernel(const int* __restrict__ EI,
                                  float* __restrict__ evu,
                                  const int* __restrict__ mkey,
                                  float* __restrict__ s, int E) {
  int e = blockIdx.x * blockDim.x + threadIdx.x;
  if (e < E) {
    int sn = EI[e];
    float ex = __expf(evu[e] - dec_f(mkey[sn]));
    evu[e] = ex;
    atomicAdd(&s[sn], ex);
  }
}
__global__ void zero_kernel(float* __restrict__ p, int n) {
  int t = blockIdx.x * blockDim.x + threadIdx.x;
  if (t < n) p[t] = 0.f;
}

// ---------------------------------------------------------------------------
// Edge block core: per 16-edge wave tile
//   avu = ex/(s[src]+1e-16); c = elu(avu * z[dst])  (f16 A tile)
//   gi  = c @ w_ih + b_ih  gate-by-gate (r -> n -> z) via WMMA
//   GRU elementwise with gh = g[dst], h = hprev[dst]; atomicAdd into out[dst]
// ---------------------------------------------------------------------------
__global__ __launch_bounds__(128) void edge_gru_kernel(
    const int* __restrict__ EI,
    const float* __restrict__ zv,      // [N][128]
    const float* __restrict__ gh,      // [N][384]
    const float* __restrict__ hprev,   // [N][128]
    const float* __restrict__ ex,      // [E] exp(evu - max)
    const float* __restrict__ ssum,    // [N]
    const half_t* __restrict__ Wih,    // packed: 24 nt x 4 kt
    const float* __restrict__ bih,     // [384]
    float* __restrict__ out,           // [N][128]
    int E) {
  __shared__ alignas(32) half_t Wlds[8 * 4 * 512];   // 32KB (one gate)
  __shared__ alignas(32) half_t Clds[4][4 * 512];    // 16KB
  __shared__ int   didxS[4][16];
  __shared__ float avuS[4][16];
  const int tid = threadIdx.x, wave = tid >> 5, lane = tid & 31;
  const int e0 = (blockIdx.x * 4 + wave) * 16;

  if (lane < 16) {
    int e = e0 + lane;
    int d = 0; float a = 0.f;
    if (e < E) {
      d = EI[E + e];
      a = ex[e] / (ssum[EI[e]] + 1e-16f);
    }
    didxS[wave][lane] = d;
    avuS[wave][lane] = a;
  }
  __syncthreads();

  for (int idx = lane; idx < 16 * 128; idx += 32) {
    int r = idx >> 7, k = idx & 127;
    int e = e0 + r;
    float v = 0.f;
    if (e < E) {
      float t = avuS[wave][r] * zv[(size_t)didxS[wave][r] * 128 + k];
      v = t > 0.f ? t : (__expf(t) - 1.f);      // ELU
    }
    Clds[wave][a_frag_off(r, k)] = (half_t)v;
  }

  const int half_id = lane >> 4, nl = lane & 15;
  v8f rr[8];                                     // holds r, then n

  for (int grp = 0; grp < 3; ++grp) {
    // gate column bases in [384]: grp0 -> r (0), grp1 -> n (256), grp2 -> z (128)
    const int ntBase = (grp == 0) ? 0 : ((grp == 1) ? 16 : 8);
    __syncthreads();
    stage_async(Wlds, Wih + (size_t)ntBase * 4 * 512, 8 * 4 * 512, tid, 128);
    __syncthreads();

    v8f gacc[8];
#pragma unroll
    for (int ntl = 0; ntl < 8; ++ntl) {
      float bv = bih[(ntBase + ntl) * 16 + nl];
      v8f acc = {bv, bv, bv, bv, bv, bv, bv, bv};
#pragma unroll
      for (int kt = 0; kt < 4; ++kt) {
        v16h a = *(const v16h*)(&Clds[wave][(kt * 32 + lane) * 16]);
        v16h b = *(const v16h*)(&Wlds[((ntl * 4 + kt) * 32 + lane) * 16]);
        acc = wmma16(a, b, acc);
      }
      gacc[ntl] = acc;
    }

#pragma unroll
    for (int ntl = 0; ntl < 8; ++ntl) {
#pragma unroll
      for (int m = 0; m < 8; ++m) {
        const int M = m + half_id * 8;
        const int row = didxS[wave][M];
        const int gcol = (ntBase + ntl) * 16 + nl;  // column in [384]
        const int hcol = ntl * 16 + nl;             // column in [128]
        float gv = gacc[ntl][m];
        if (grp == 0) {
          rr[ntl][m] = sigm(gv + gh[(size_t)row * 384 + gcol]);          // r
        } else if (grp == 1) {
          rr[ntl][m] = tanhf(gv + rr[ntl][m] * gh[(size_t)row * 384 + gcol]); // n
        } else {
          float z = sigm(gv + gh[(size_t)row * 384 + gcol]);
          float h = (1.f - z) * rr[ntl][m] + z * hprev[(size_t)row * 128 + hcol];
          if (e0 + M < E) atomicAdd(&out[(size_t)row * 128 + hcol], h);
        }
      }
    }
  }
}

// ---------------------------------------------------------------------------
extern "C" void kernel_launch(void* const* d_in, const int* in_sizes, int n_in,
                              void* d_out, int out_size, void* d_ws, size_t ws_size,
                              hipStream_t stream) {
  const int N = in_sizes[0] / 39;
  const int E = in_sizes[1] / 2;

  const float* X   = (const float*)d_in[0];
  const int*   EI  = (const int*)d_in[1];
  const float* EA  = (const float*)d_in[2];
  const float* atomW  = (const float*)d_in[3];
  const float* atomB  = (const float*)d_in[4];
  const float* neighW = (const float*)d_in[5];
  const float* neighB = (const float*)d_in[6];
  const float* al0W = (const float*)d_in[7];
  const float* al0B = (const float*)d_in[8];
  const float* at0W = (const float*)d_in[9];
  const float* at0B = (const float*)d_in[10];
  const float* wih0 = (const float*)d_in[11];
  const float* whh0 = (const float*)d_in[12];
  const float* bih0 = (const float*)d_in[13];
  const float* bhh0 = (const float*)d_in[14];
  const float* al1W = (const float*)d_in[15];
  const float* al1B = (const float*)d_in[16];
  const float* at1W = (const float*)d_in[17];
  const float* at1B = (const float*)d_in[18];
  const float* wih1 = (const float*)d_in[19];
  const float* whh1 = (const float*)d_in[20];
  const float* bih1 = (const float*)d_in[21];
  const float* bhh1 = (const float*)d_in[22];

  char* ws = (char*)d_ws;
  size_t off = 0;
  auto alloc = [&](size_t bytes) -> void* {
    void* p = ws + off;
    off = (off + bytes + 255) & ~(size_t)255;
    return p;
  };
  // packed f16 weights (fragment order)
  half_t* pAtom  = (half_t*)alloc(8  * 2 * 512 * 2);   // K64  x N128
  half_t* pNeigh = (half_t*)alloc(8  * 2 * 512 * 2);   // K64  x N128
  half_t* pAtt0  = (half_t*)alloc(8  * 4 * 512 * 2);   // K128 x N128
  half_t* pAtt1  = (half_t*)alloc(8  * 4 * 512 * 2);
  half_t* pIh0   = (half_t*)alloc(24 * 4 * 512 * 2);   // K128 x N384
  half_t* pHh0   = (half_t*)alloc(24 * 4 * 512 * 2);
  half_t* pIh1   = (half_t*)alloc(24 * 4 * 512 * 2);
  half_t* pHh1   = (half_t*)alloc(24 * 4 * 512 * 2);
  // node/edge intermediates
  float* y    = (float*)alloc((size_t)N * 128 * 4);
  float* x1   = (float*)alloc((size_t)N * 128 * 4);
  float* z    = (float*)alloc((size_t)N * 128 * 4);
  float* g    = (float*)alloc((size_t)N * 384 * 4);
  float* p    = (float*)alloc((size_t)N * 4);
  float* q    = (float*)alloc((size_t)N * 4);
  int*   mkey = (int*)  alloc((size_t)N * 4);
  float* ssum = (float*)alloc((size_t)N * 4);
  float* evu  = (float*)alloc((size_t)E * 4);
  float* outF = (float*)d_out;

  auto cdiv = [](int a, int b) { return (a + b - 1) / b; };
  const int gN  = cdiv(N, 64);        // node_gemm grid
  const int gE  = cdiv(E, 64);        // edge-tile grid
  const int tE  = cdiv(E, 256);
  const int tN  = cdiv(N, 256);
  const int tNF = cdiv(N * 128, 256);

  // ---- pack weights ----
  auto pack = [&](const float* w, half_t* o, int Kin, int Kpad, int Nout) {
    int total = (Nout >> 4) * (Kpad >> 5) * 512;
    pack_weight_kernel<<<cdiv(total, 256), 256, 0, stream>>>(w, o, Kin, Kpad, Nout);
  };
  pack(atomW,  pAtom,  39, 64, 128);
  pack(neighW, pNeigh, 49, 64, 128);
  pack(at0W,   pAtt0, 128, 128, 128);
  pack(at1W,   pAtt1, 128, 128, 128);
  pack(wih0,   pIh0,  128, 128, 384);
  pack(whh0,   pHh0,  128, 128, 384);
  pack(wih1,   pIh1,  128, 128, 384);
  pack(whh1,   pHh1,  128, 128, 384);

  // ---- layer 0 node-side ----
  node_gemm_kernel<<<gN, 128, 0, stream>>>(X, 39, 64, pAtom, 128, atomB, y, N, 1);
  node_gemm_kernel<<<gN, 128, 0, stream>>>(y, 128, 128, pAtt0, 128, at0B, z, N, 0);
  node_gemm_kernel<<<gN, 128, 0, stream>>>(y, 128, 128, pHh0, 384, bhh0, g, N, 0);
  node_dot_kernel<<<cdiv(N, 8), 256, 0, stream>>>(y, al0W, 128, p, N);

  // ---- layer 0 edge-side ----
  edge_evu_l0_kernel<<<gE, 128, 0, stream>>>(X, EI, EA, pNeigh, neighB,
                                             al0W, al0B, p, evu, E);
  init_seg_kernel<<<tN, 256, 0, stream>>>(mkey, ssum, N);
  seg_max_kernel<<<tE, 256, 0, stream>>>(EI, evu, mkey, E);
  seg_expsum_kernel<<<tE, 256, 0, stream>>>(EI, evu, mkey, ssum, E);
  zero_kernel<<<tNF, 256, 0, stream>>>(x1, N * 128);
  edge_gru_kernel<<<gE, 128, 0, stream>>>(EI, z, g, y, evu, ssum, pIh0, bih0, x1, E);

  // ---- layer 1 node-side ----
  node_gemm_kernel<<<gN, 128, 0, stream>>>(x1, 128, 128, pAtt1, 128, at1B, z, N, 0);
  node_gemm_kernel<<<gN, 128, 0, stream>>>(x1, 128, 128, pHh1, 384, bhh1, g, N, 0);
  node_dot_kernel<<<cdiv(N, 8), 256, 0, stream>>>(x1, al1W, 128, p, N);
  node_dot_kernel<<<cdiv(N, 8), 256, 0, stream>>>(x1, al1W + 128, 128, q, N);

  // ---- layer 1 edge-side ----
  edge_evu_l1_kernel<<<tE, 256, 0, stream>>>(EI, p, q, al1B, evu, E);
  init_seg_kernel<<<tN, 256, 0, stream>>>(mkey, ssum, N);
  seg_max_kernel<<<tE, 256, 0, stream>>>(EI, evu, mkey, E);
  seg_expsum_kernel<<<tE, 256, 0, stream>>>(EI, evu, mkey, ssum, E);
  zero_kernel<<<tNF, 256, 0, stream>>>(outF, N * 128);
  edge_gru_kernel<<<gE, 128, 0, stream>>>(EI, z, g, x1, evu, ssum, pIh1, bih1, outF, E);

  (void)n_in; (void)out_size; (void)ws_size;
}
